// OTAlignment_8383776161865
// MI455X (gfx1250) — matrix operand order repaired and checked by reading, not compiled, not run on hardware
//
#include <hip/hip_runtime.h>
#include <stdint.h>

// ---------------------------------------------------------------------------
// OTAlignment on MI455X (gfx1250).
//
// Sinkhorn collapses analytically: log(1f+1e-8) == 0 in fp32, so the
// transport scalar P == exp(0) == 1.0 for every sample; the cost MLP is dead.
// Output is exactly:
//   out = [ img@W_img+b_img | eds@W_eds+b_eds | xrd@W_xrd+b_xrd ]  (B x 1536)
//
// Three bias-fused bf16-WMMA GEMMs (f32 accumulate). Memory roofline:
// ~740 MB unique traffic -> ~32 us at 23.3 TB/s. Design:
//   * N tile is the fast grid dimension: the 4 blocks sharing an A row-panel
//     run concurrently so the 192 MB L2 dedupes the panel (no 4x HBM re-read),
//   * double-buffered LDS, one barrier per K-step, next tile's global loads
//     issued before the WMMAs so load latency hides under matrix math,
//   * LDS row stride of 20 uints: 16B-aligned b128 DS accesses that are
//     provably bank-conflict-free (20*m mod 64, m=0..15, covers each of the
//     64 banks exactly once per half-wave),
//   * all LDS addressing by integer offset into the __shared__ arrays so the
//     compiler emits ds_* ops (a pointer array here previously decayed to
//     generic/flat stores),
//   * fp32->bf16 pack = 2 adds + 1 v_perm per pair, done once at staging.
// ---------------------------------------------------------------------------

typedef __attribute__((ext_vector_type(16))) __bf16 v16bf;
typedef __attribute__((ext_vector_type(2)))  __bf16 v2bf;
typedef __attribute__((ext_vector_type(8)))  float  v8f;

#define TILE_M 128
#define TILE_N 128
#define TILE_K 32
#define LDS_STRIDE 20   // uints per row: 16 data + 4 pad -> 16B aligned, 0-conflict
#define OUT_STRIDE 1536

#if defined(__has_builtin)
#  if __has_builtin(__builtin_amdgcn_cvt_pk_bf16_f32)
#    define HAVE_CVT_PK_BF16 1
#  endif
#endif

// Pack two fp32 -> one uint32 of two bf16 (a -> low 16, b -> high 16).
__device__ __forceinline__ uint32_t pack2_bf16(float a, float b) {
#ifdef HAVE_CVT_PK_BF16
    v2bf r = __builtin_amdgcn_cvt_pk_bf16_f32(a, b);
    return __builtin_bit_cast(uint32_t, r);
#else
    // round-half-up then byte-select the two high halves with v_perm_b32
    uint32_t ua = __builtin_bit_cast(uint32_t, a) + 0x8000u;
    uint32_t ub = __builtin_bit_cast(uint32_t, b) + 0x8000u;
    return __builtin_amdgcn_perm(ub, ua, 0x07060302u);
#endif
}

union FragCast { uint4 q[2]; v16bf v; };

// A fragment (16x32 bf16, MxK), ISA 7.12.2 16-bit A layout:
// lanes 0-15 hold K 0-7 (VGPR0-3) & 16-23 (VGPR4-7); lanes 16-31 K 8-15 & 24-31.
// Two aligned b128 LDS loads per fragment.
__device__ __forceinline__ v16bf load_frag_a(const uint32_t* __restrict__ rowBase, int lane) {
    const int m  = lane & 15;
    const int hi = (lane >> 4) & 1;
    const uint32_t* r = rowBase + m * LDS_STRIDE + hi * 4;
    FragCast f;
    f.q[0] = *reinterpret_cast<const uint4*>(r);
    f.q[1] = *reinterpret_cast<const uint4*>(r + 8);
    return f.v;
}

// B fragment (32x16 bf16, KxN), stored N-major (W^T rows) in LDS:
// lanes 0-15 hold K 0-7 & 16-23, lanes 16-31 hold K 8-15 & 24-31 (matches A).
__device__ __forceinline__ v16bf load_frag_b(const uint32_t* __restrict__ rowBase, int lane) {
    const int n  = lane & 15;
    const int hi = (lane >> 4) & 1;
    const uint32_t* r = rowBase + n * LDS_STRIDE + hi * 4;
    FragCast f;
    f.q[0] = *reinterpret_cast<const uint4*>(r);
    f.q[1] = *reinterpret_cast<const uint4*>(r + 8);
    return f.v;
}

__global__ __launch_bounds__(256)
void gemm_bias_wmma_bf16(const float* __restrict__ A,     // [B, K] fp32
                         const float* __restrict__ W,     // [K, 512] fp32
                         const float* __restrict__ bias,  // [512] fp32
                         float* __restrict__ out,         // 512-col slice base
                         int K, float scale) {
    __shared__ uint32_t sA[2][TILE_M * LDS_STRIDE];
    __shared__ uint32_t sB[2][TILE_N * LDS_STRIDE];

    const int tid  = threadIdx.x;
    const int lane = tid & 31;
    const int wave = tid >> 5;                 // 0..7
    const int wm   = (wave & 3) * 32;          // wave M offset in tile
    const int wn   = (wave >> 2) * 64;         // wave N offset in tile

    const int n0 = blockIdx.x * TILE_N;        // FAST dim -> A panel L2 reuse
    const int m0 = blockIdx.y * TILE_M;

    // Stager assignment (256 threads): thread -> (row 0..127, 16-elem half)
    const int sr = tid >> 1;
    const int sh = tid & 1;
    const int stOff = sr * LDS_STRIDE + sh * 8;   // uint offset; 16B aligned

    const float* aSrc = A + (size_t)(m0 + sr) * K + sh * 16;        // A[m][k..]
    const float* wSrc = W + (size_t)(sh * 16) * 512 + n0 + sr;      // W[k][n]

    v8f acc[2][4];
#pragma unroll
    for (int i = 0; i < 2; ++i)
#pragma unroll
        for (int j = 0; j < 4; ++j) acc[i][j] = v8f{};

    const int nk = K / TILE_K;

    float4 ga[4];
    float  gw[16];

    // ---- prologue: fetch + stage tile 0 into buffer 0 ----------------------
    {
        const float4* ap = reinterpret_cast<const float4*>(aSrc);
#pragma unroll
        for (int i = 0; i < 4; ++i) ga[i] = ap[i];
#pragma unroll
        for (int i = 0; i < 16; ++i) gw[i] = wSrc[(size_t)i * 512];

        uint4 a0 = { pack2_bf16(ga[0].x, ga[0].y), pack2_bf16(ga[0].z, ga[0].w),
                     pack2_bf16(ga[1].x, ga[1].y), pack2_bf16(ga[1].z, ga[1].w) };
        uint4 a1 = { pack2_bf16(ga[2].x, ga[2].y), pack2_bf16(ga[2].z, ga[2].w),
                     pack2_bf16(ga[3].x, ga[3].y), pack2_bf16(ga[3].z, ga[3].w) };
        uint4* ad = reinterpret_cast<uint4*>(&sA[0][stOff]);
        ad[0] = a0; ad[1] = a1;

        uint4 b0 = { pack2_bf16(gw[0],  gw[1]),  pack2_bf16(gw[2],  gw[3]),
                     pack2_bf16(gw[4],  gw[5]),  pack2_bf16(gw[6],  gw[7]) };
        uint4 b1 = { pack2_bf16(gw[8],  gw[9]),  pack2_bf16(gw[10], gw[11]),
                     pack2_bf16(gw[12], gw[13]), pack2_bf16(gw[14], gw[15]) };
        uint4* bd = reinterpret_cast<uint4*>(&sB[0][stOff]);
        bd[0] = b0; bd[1] = b1;
    }
    __syncthreads();

    // ---- main loop: 1 barrier per K-step, loads overlap WMMAs --------------
    for (int kt = 0; kt < nk; ++kt) {
        const int  cur  = kt & 1;
        const bool more = (kt + 1) < nk;

        if (more) {  // issue next tile's global loads before computing
            const float4* ap =
                reinterpret_cast<const float4*>(aSrc + (size_t)(kt + 1) * TILE_K);
#pragma unroll
            for (int i = 0; i < 4; ++i) ga[i] = ap[i];
            const float* wp = wSrc + (size_t)(kt + 1) * TILE_K * 512;
#pragma unroll
            for (int i = 0; i < 16; ++i) gw[i] = wp[(size_t)i * 512];
            if (kt + 2 < nk)
                __builtin_prefetch(aSrc + (size_t)(kt + 2) * TILE_K, 0, 0);
        }

        // compute on buffer `cur`
        v16bf afr[2], bfr[4];
#pragma unroll
        for (int i = 0; i < 2; ++i)
            afr[i] = load_frag_a(&sA[cur][(wm + 16 * i) * LDS_STRIDE], lane);
#pragma unroll
        for (int j = 0; j < 4; ++j)
            bfr[j] = load_frag_b(&sB[cur][(wn + 16 * j) * LDS_STRIDE], lane);

#pragma unroll
        for (int i = 0; i < 2; ++i)
#pragma unroll
            for (int j = 0; j < 4; ++j)
                acc[i][j] = __builtin_amdgcn_wmma_f32_16x16x32_bf16(
                    false, afr[i], false, bfr[j],
                    (short)0, acc[i][j], false, false);

        if (more) {  // stage next tile into the other buffer (pure DS stores)
            const int nxt = cur ^ 1;
            uint4 a0 = { pack2_bf16(ga[0].x, ga[0].y), pack2_bf16(ga[0].z, ga[0].w),
                         pack2_bf16(ga[1].x, ga[1].y), pack2_bf16(ga[1].z, ga[1].w) };
            uint4 a1 = { pack2_bf16(ga[2].x, ga[2].y), pack2_bf16(ga[2].z, ga[2].w),
                         pack2_bf16(ga[3].x, ga[3].y), pack2_bf16(ga[3].z, ga[3].w) };
            uint4* ad = reinterpret_cast<uint4*>(&sA[nxt][stOff]);
            ad[0] = a0; ad[1] = a1;

            uint4 b0 = { pack2_bf16(gw[0],  gw[1]),  pack2_bf16(gw[2],  gw[3]),
                         pack2_bf16(gw[4],  gw[5]),  pack2_bf16(gw[6],  gw[7]) };
            uint4 b1 = { pack2_bf16(gw[8],  gw[9]),  pack2_bf16(gw[10], gw[11]),
                         pack2_bf16(gw[12], gw[13]), pack2_bf16(gw[14], gw[15]) };
            uint4* bd = reinterpret_cast<uint4*>(&sB[nxt][stOff]);
            bd[0] = b0; bd[1] = b1;
        }
        __syncthreads();
    }

    // ---- epilogue: bias add (+scale), fp32 store ---------------------------
    // C/D layout: lane = hi*16 + n ; VGPR r holds row m = r + hi*8.
    const int cn  = lane & 15;
    const int chi = lane >> 4;
#pragma unroll
    for (int i = 0; i < 2; ++i) {
#pragma unroll
        for (int j = 0; j < 4; ++j) {
            const int   nglob = n0 + wn + 16 * j + cn;     // 0..511
            const float bv    = bias[nglob];
            const int   mbase = m0 + wm + 16 * i + chi * 8;
            float* op = out + (size_t)mbase * OUT_STRIDE + nglob;
#pragma unroll
            for (int r = 0; r < 8; ++r)
                op[(size_t)r * OUT_STRIDE] = (acc[i][j][r] + bv) * scale;
        }
    }
}

extern "C" void kernel_launch(void* const* d_in, const int* in_sizes, int n_in,
                              void* d_out, int out_size, void* d_ws, size_t ws_size,
                              hipStream_t stream) {
    (void)n_in; (void)out_size; (void)d_ws; (void)ws_size;

    const float* img   = (const float*)d_in[0];   // [B, 768]
    const float* eds   = (const float*)d_in[1];   // [B, 256]
    const float* xrd   = (const float*)d_in[2];   // [B, 256]
    const float* W_img = (const float*)d_in[3];   // [768, 512]
    const float* b_img = (const float*)d_in[4];   // [512]
    const float* W_eds = (const float*)d_in[5];   // [256, 512]
    const float* b_eds = (const float*)d_in[6];   // [512]
    const float* W_xrd = (const float*)d_in[7];   // [256, 512]
    const float* b_xrd = (const float*)d_in[8];   // [512]
    // d_in[9..12] (cost MLP) are provably dead: Sinkhorn transport == 1.0f.

    float* out = (float*)d_out;                   // [B, 1536]
    const int B = in_sizes[0] / 768;              // 65536

    dim3 block(256);
    dim3 grid(512 / TILE_N, B / TILE_M);          // n fast (L2 panel reuse), m slow

    gemm_bias_wmma_bf16<<<grid, block, 0, stream>>>(img, W_img, b_img, out +    0, 768, 1.0f);
    gemm_bias_wmma_bf16<<<grid, block, 0, stream>>>(eds, W_eds, b_eds, out +  512, 256, 1.0f);
    gemm_bias_wmma_bf16<<<grid, block, 0, stream>>>(xrd, W_xrd, b_xrd, out + 1024, 256, 1.0f);
}
